// Linear_687194768080
// MI455X (gfx1250) — compile-verified, matrix operand order checked by reading, not run on hardware
//
#include <hip/hip_runtime.h>

// ---------------------------------------------------------------------------
// Quantized linear: out[32,16384] = x[32,4096] @ (w_i8 * scale[o])^T
// HBM-bound on the 268 MB int32 weight stream (floor ~11.5 us @ 23.3 TB/s);
// compute via v_wmma_f32_16x16x32_f16 (weight codes [-127,127] exact in f16,
// f32 accumulation). Each wave owns a 32x16 output tile so every weight
// element is read from HBM exactly once.
// ---------------------------------------------------------------------------

typedef __attribute__((ext_vector_type(16))) _Float16 v16h;
typedef __attribute__((ext_vector_type(8)))  _Float16 v8h;
typedef __attribute__((ext_vector_type(4)))  _Float16 v4h;
typedef __attribute__((ext_vector_type(8)))  float    v8f;
typedef __attribute__((ext_vector_type(4)))  float    v4f;
typedef __attribute__((ext_vector_type(4)))  int      v4i;

#define K_DIM   4096
#define N_DIM   16384
#define M_DIM   32
#define BK      512               // K-chunk staged in LDS
#define LDSS    (BK + 8)          // padded row stride (halves): 4-bank skew/row
#define NTHREADS 128              // 4 waves/block -> 256 blocks (WGP coverage)
#define NWAVES  (NTHREADS / 32)

__global__ __launch_bounds__(NTHREADS)
void qlinear_wmma_f16(const float* __restrict__ x,
                      const int*   __restrict__ w,
                      const float* __restrict__ scaler,
                      float*       __restrict__ out)
{
    // x staged as f16: 32 rows x (512+8) halves = 33,280 B LDS
    __shared__ _Float16 lds_x[M_DIM * LDSS];

    const int tid  = threadIdx.x;
    const int wave = tid >> 5;
    const int lane = tid & 31;
    const int n    = lane & 15;   // N within tile  (also M for A frags)
    const int hp   = lane >> 4;   // K-half selector per ISA 16-bit A/B layout

    // each wave owns one 16-wide N tile, all 32 M rows
    const int col0 = (blockIdx.x * NWAVES + wave) * 16;
    const int* __restrict__ wrow = w + (size_t)(col0 + n) * K_DIM;

    v8f acc0 = {};  // M = 0..15
    v8f acc1 = {};  // M = 16..31

    for (int c = 0; c < K_DIM / BK; ++c) {
        // ---- stage x chunk [32][BK] fp32 -> f16 into LDS ----
        __syncthreads();                       // protect LDS WAR across chunks
        {
            const float* __restrict__ xsrc = x + c * BK;
            #pragma unroll
            for (int i = 0; i < (M_DIM * BK) / (NTHREADS * 4); ++i) {
                int e   = (tid + i * NTHREADS) * 4;
                int row = e >> 9;              // / BK
                int kl  = e & (BK - 1);
                v4f v = *(const v4f*)(xsrc + (size_t)row * K_DIM + kl);
                v4h h = { (_Float16)v.x, (_Float16)v.y,
                          (_Float16)v.z, (_Float16)v.w };
                *(v4h*)(&lds_x[row * LDSS + kl]) = h;
            }
        }
        __syncthreads();

        const int kgbl = c * BK;
        #pragma unroll 4
        for (int kk = 0; kk < BK; kk += 32) {
            // ---- B fragment: 16 int32 codes -> f16 (exact for [-127,127]) ----
            // lane covers w[col0+n][k + hp*8 .. +7] and [k+16+hp*8 .. +7];
            // 32 lanes together = 16 rows x 128 B = 16 full cachelines.
            const int* wp = wrow + kgbl + kk + hp * 8;
            v4i b0 = __builtin_nontemporal_load((const v4i*)(wp +  0));
            v4i b1 = __builtin_nontemporal_load((const v4i*)(wp +  4));
            v4i b2 = __builtin_nontemporal_load((const v4i*)(wp + 16));
            v4i b3 = __builtin_nontemporal_load((const v4i*)(wp + 20));
            v16h bf;
            bf[0]  = (_Float16)b0.x; bf[1]  = (_Float16)b0.y;
            bf[2]  = (_Float16)b0.z; bf[3]  = (_Float16)b0.w;
            bf[4]  = (_Float16)b1.x; bf[5]  = (_Float16)b1.y;
            bf[6]  = (_Float16)b1.z; bf[7]  = (_Float16)b1.w;
            bf[8]  = (_Float16)b2.x; bf[9]  = (_Float16)b2.y;
            bf[10] = (_Float16)b2.z; bf[11] = (_Float16)b2.w;
            bf[12] = (_Float16)b3.x; bf[13] = (_Float16)b3.y;
            bf[14] = (_Float16)b3.z; bf[15] = (_Float16)b3.w;

            // ---- A fragments from LDS (16-B ds loads, conflict-free pad) ----
            const _Float16* lp = &lds_x[kk + hp * 8];
            v8h a0lo = *(const v8h*)(lp + n * LDSS);
            v8h a0hi = *(const v8h*)(lp + n * LDSS + 16);
            v8h a1lo = *(const v8h*)(lp + (n + 16) * LDSS);
            v8h a1hi = *(const v8h*)(lp + (n + 16) * LDSS + 16);
            v16h af0 = __builtin_shufflevector(a0lo, a0hi,
                        0,1,2,3,4,5,6,7,8,9,10,11,12,13,14,15);
            v16h af1 = __builtin_shufflevector(a1lo, a1hi,
                        0,1,2,3,4,5,6,7,8,9,10,11,12,13,14,15);

            // D = A x B + C   (8-arg form: neg_a, A, neg_b, B, c_mod, C, reuse)
            acc0 = __builtin_amdgcn_wmma_f32_16x16x32_f16(
                       false, af0, false, bf, (short)0, acc0, false, false);
            acc1 = __builtin_amdgcn_wmma_f32_16x16x32_f16(
                       false, af1, false, bf, (short)0, acc1, false, false);
        }
    }

    // ---- epilogue: per-output-channel scale, scatter per C/D layout ----
    const float s = scaler[col0 + n];
    float* __restrict__ op = out + col0 + n;
    #pragma unroll
    for (int r = 0; r < 8; ++r) {
        int m0 = r + hp * 8;                       // VGPR r -> M = r + 8*(lane/16)
        op[(size_t)m0 * N_DIM]        = acc0[r] * s;
        op[(size_t)(m0 + 16) * N_DIM] = acc1[r] * s;
    }
}

extern "C" void kernel_launch(void* const* d_in, const int* in_sizes, int n_in,
                              void* d_out, int out_size, void* d_ws, size_t ws_size,
                              hipStream_t stream)
{
    const float* x  = (const float*)d_in[0];   // [1,32,4096] fp32
    const int*   w  = (const int*)  d_in[1];   // [16384,4096] int32 codes
    const float* sc = (const float*)d_in[2];   // [16384] fp32
    float*       o  = (float*)d_out;           // [1,32,16384] fp32

    // 1024 N-tiles of 16 cols; 4 waves/block -> 256 blocks x 128 threads
    qlinear_wmma_f16<<<dim3(N_DIM / (16 * NWAVES)), dim3(NTHREADS), 0, stream>>>(
        x, w, sc, o);
}
